// MultiHeadSelfAttention_41472204210244
// MI455X (gfx1250) — matrix operand order repaired and checked by reading, not compiled, not run on hardware
//
#include <hip/hip_runtime.h>

// ---------------------------------------------------------------------------
// MultiHeadSelfAttention for MI455X (gfx1250, wave32, WMMA, TDM)
//   B=4, S=2048, D=1024, H=16, HD=64
//   Stage 1: fp32 -> bf16 converts (x, Wq, Wk, Wv)
//   Stage 2: QKV projection as cooperative LDS-tiled GEMM (128x128 block
//            tile), tiles staged by the Tensor Data Mover (tensor_load_to_lds,
//            TENSORcnt, double-buffered) with padded LDS rows (32 -> 40 u16)
//   Stage 3: fused flash-attention (online softmax), 8 WMMA per 32-key block
// ---------------------------------------------------------------------------

typedef __bf16 v16bf __attribute__((ext_vector_type(16)));
typedef float  v8f   __attribute__((ext_vector_type(8)));
typedef unsigned int v4u __attribute__((ext_vector_type(4)));
typedef int v4i __attribute__((ext_vector_type(4)));
typedef int v8i __attribute__((ext_vector_type(8)));

#if __has_builtin(__builtin_amdgcn_tensor_load_to_lds) && \
    __has_builtin(__builtin_amdgcn_s_wait_tensorcnt)
#define HAVE_TDM 1
#else
#define HAVE_TDM 0
#endif

union FragAB {
  v16bf v;
  uint4 q[2];
  unsigned short s[16];
};

__device__ __forceinline__ unsigned short f2bf(float x) {
  union { float f; unsigned u; } v; v.f = x;
  unsigned u = v.u;
  u += 0x7FFFu + ((u >> 16) & 1u);       // round to nearest even
  return (unsigned short)(u >> 16);
}

// ---- fragment loaders, global memory (row-major, leading dim ld) ----------
// A 16x32 bf16 (ISA 7.12.2): lane m=lane&15,h=lane>>4;
//   elems 0..7 = A[m, k0+h*8+j]; elems 8..15 = A[m, k0+16+h*8+j]
__device__ __forceinline__ v16bf load_frag_a(const unsigned short* __restrict__ base,
                                             int row0, int ld, int k0, int lane) {
  const int m = lane & 15, h = lane >> 4;
  const unsigned short* r = base + (size_t)(row0 + m) * ld + k0 + h * 8;
  FragAB f;
  f.q[0] = *(const uint4*)(r);
  f.q[1] = *(const uint4*)(r + 16);
  return f.v;
}
// B 32x16 bf16, B(k,n) = src[n0+n, k0+k]: lane n=lane&15,h=lane>>4;
//   elems 0..7 = src[n0+n, k0+h*8+j]; elems 8..15 = src[n0+n, k0+16+h*8+j]
__device__ __forceinline__ v16bf load_frag_b(const unsigned short* __restrict__ base,
                                             int n0, int ld, int k0, int lane) {
  const int n = lane & 15, h = lane >> 4;
  const unsigned short* r = base + (size_t)(n0 + n) * ld + k0 + h * 8;
  FragAB f;
  f.q[0] = *(const uint4*)(r);
  f.q[1] = *(const uint4*)(r + 16);
  return f.v;
}

// ---- fragment loaders, LDS tiles with padded rows (stride 40 u16 = 80 B) --
__device__ __forceinline__ v16bf load_frag_lds_a(const unsigned short* base,
                                                 int row0, int lane) {
  const int m = lane & 15, h = lane >> 4;
  const unsigned short* r = base + (row0 + m) * 40 + h * 8;
  FragAB f;
  f.q[0] = *(const uint4*)(r);
  f.q[1] = *(const uint4*)(r + 16);
  return f.v;
}
__device__ __forceinline__ v16bf load_frag_lds_b(const unsigned short* base,
                                                 int n0, int lane) {
  const int n = lane & 15, h = lane >> 4;
  const unsigned short* r = base + (n0 + n) * 40 + h * 8;
  FragAB f;
  f.q[0] = *(const uint4*)(r);
  f.q[1] = *(const uint4*)(r + 16);
  return f.v;
}

// ---- TDM: DMA a 2D bf16 tile (tile_rows x tile_k) global -> LDS -----------
// Pads each 32-element (16-DWORD) row with 4 DWORDs -> LDS row stride 40 u16.
__device__ __forceinline__ void tdm_load_tile(unsigned lds_byte_off,
                                              const void* gaddr,
                                              unsigned tile_k, unsigned tile_rows,
                                              unsigned row_stride_elems) {
#if HAVE_TDM
  const unsigned long long ga = (unsigned long long)(uintptr_t)gaddr;
  v4u g0;
  g0[0] = 1u;                                             // count=1, user D#
  g0[1] = lds_byte_off;                                   // LDS byte address
  g0[2] = (unsigned)(ga & 0xFFFFFFFFu);                   // global_addr[31:0]
  g0[3] = (unsigned)((ga >> 32) & 0x1FFFFFFu) | (2u << 30); // addr[56:32]|type=2
  v8i g1;
  g1[0] = (int)((1u << 16)        // data_size = 2 bytes
              | (1u << 20)        // pad_enable
              | (3u << 22)        // pad_interval: 16 DWORDs (one tile row)
              | (3u << 25));      // pad_amount: 4 DWORDs (8 bf16)
  g1[1] = (int)(0x4000u << 16);   // tensor_dim0[15:0] = 16384
  g1[2] = (int)(0x4000u << 16);   // tensor_dim0[31:16]=0 | tensor_dim1[15:0]=16384
  g1[3] = (int)(tile_k << 16);    // tensor_dim1[31:16]=0 | tile_dim0
  g1[4] = (int)(tile_rows & 0xFFFFu); // tile_dim1 | tile_dim2=0 (2D)
  g1[5] = (int)row_stride_elems;  // tensor_dim0_stride[31:0]
  g1[6] = 0;                      // stride0[47:32]=0 | dim1_stride[15:0]=0
  g1[7] = 0;
  const v4i z4 = {0, 0, 0, 0};              // VADDR2/3 groups unused (2D)
  const v8i z8 = {0, 0, 0, 0, 0, 0, 0, 0};  // 5th group (6-arg toolchain)
  __builtin_amdgcn_tensor_load_to_lds(g0, g1, z4, z4, z8, 0);
#endif
}

// ---------------------------------------------------------------------------
__global__ void __launch_bounds__(256)
cvt_f32_bf16(const float* __restrict__ in, unsigned short* __restrict__ out, int n) {
  int i = blockIdx.x * blockDim.x + threadIdx.x;
  const int stride = gridDim.x * blockDim.x;
  for (; i < n; i += stride) out[i] = f2bf(in[i]);
}

// ---------------------------------------------------------------------------
// QKV projection: out[n,o] = sum_d xb[n,d]*W[o,d] + bias[o]
// Block (8 waves) -> 128(M) x 128(N) tile; wave -> 32(M) x 64(N).
// K-loop step 32, double-buffered TDM staging of A(128x32) and W(128x32).
// Grid: 3 * (8192/128) * (1024/128) = 1536 blocks.
__global__ void __launch_bounds__(256)
qkv_proj(const unsigned short* __restrict__ xb,
         const unsigned short* __restrict__ wb,   // 3 x [1024][1024] bf16
         const float* __restrict__ bq, const float* __restrict__ bk,
         const float* __restrict__ bv,
         unsigned short* __restrict__ Qo,         // [8192][1024]
         unsigned short* __restrict__ Ko,         // [8192][1024]
         unsigned short* __restrict__ Vt)         // [64][64][2048] per-head ^T
{
  // [A0 5120][B0 5120][A1 5120][B1 5120] u16 (rows padded 32->40) = 40 KB
  __shared__ __align__(16) unsigned short lds[4 * 5120];

  const int tid  = threadIdx.x;
  const int lane = tid & 31;
  const int w    = tid >> 5;

  const int bid  = blockIdx.x;           // ((wsel*64)+mt)*8 + nt
  const int nt   = bid & 7;
  const int mt   = (bid >> 3) & 63;
  const int wsel = bid >> 9;             // 0=Q,1=K,2=V
  const int m0   = mt * 128;
  const int n0   = nt * 128;

  const unsigned short* W = wb + (size_t)wsel * (1024u * 1024u);
  const float* bias = (wsel == 0) ? bq : (wsel == 1 ? bk : bv);

  const unsigned ldsbase = (unsigned)(uintptr_t)&lds[0];
  const int wm = w >> 1;                 // 0..3 -> M offset wm*32
  const int wn = w & 1;                  // 0..1 -> N offset wn*64

  v8f acc[2][4] = {};

  auto issue = [&](int buf, int k0) {
#if HAVE_TDM
    if (w == 0) {
      tdm_load_tile(ldsbase + (unsigned)buf * 20480u,
                    xb + (size_t)m0 * 1024 + k0, 32u, 128u, 1024u);
      tdm_load_tile(ldsbase + (unsigned)buf * 20480u + 10240u,
                    W + (size_t)n0 * 1024 + k0, 32u, 128u, 1024u);
    }
#else
    unsigned short* la = lds + buf * 10240;
    unsigned short* lb = la + 5120;
    const unsigned short* ga = xb + (size_t)m0 * 1024 + k0;
    const unsigned short* gb = W + (size_t)n0 * 1024 + k0;
#pragma unroll
    for (int t = 0; t < 2; ++t) {        // 512 8-elem chunks / 256 threads
      const int c = tid + t * 256;
      const int row = c >> 2, cc = (c & 3) * 8;
      *(uint4*)(la + row * 40 + cc) = *(const uint4*)(ga + (size_t)row * 1024 + cc);
      *(uint4*)(lb + row * 40 + cc) = *(const uint4*)(gb + (size_t)row * 1024 + cc);
    }
#endif
  };

  issue(0, 0);
  for (int kt = 0; kt < 32; ++kt) {
#if HAVE_TDM
    if (w == 0) __builtin_amdgcn_s_wait_tensorcnt(0);
#endif
    __syncthreads();                              // tile (kt&1) ready block-wide
    if (kt + 1 < 32) issue((kt + 1) & 1, (kt + 1) * 32);  // overlap next DMA

    const unsigned short* la = lds + (kt & 1) * 10240;
    const unsigned short* lb = la + 5120;
    const v16bf a0 = load_frag_lds_a(la, wm * 32,      lane);
    const v16bf a1 = load_frag_lds_a(la, wm * 32 + 16, lane);
#pragma unroll
    for (int j = 0; j < 4; ++j) {
      const v16bf b = load_frag_lds_b(lb, wn * 64 + j * 16, lane);
      acc[0][j] = __builtin_amdgcn_wmma_f32_16x16x32_bf16(
                      false, a0, false, b, (short)0, acc[0][j], false, false);
      acc[1][j] = __builtin_amdgcn_wmma_f32_16x16x32_bf16(
                      false, a1, false, b, (short)0, acc[1][j], false, false);
    }
  }

  // ---- epilogue: bias add, bf16 convert, scatter stores --------------------
  // C layout: lane col n=lane&15, rows m = r + 8*(lane>>4)
  const int nlane = lane & 15, h = lane >> 4;
#pragma unroll
  for (int mi = 0; mi < 2; ++mi) {
#pragma unroll
    for (int j = 0; j < 4; ++j) {
      const int o = n0 + wn * 64 + j * 16 + nlane;
      const float bo = bias[o];
      if (wsel < 2) {
        unsigned short* out = (wsel == 0) ? Qo : Ko;
#pragma unroll
        for (int r = 0; r < 8; ++r) {
          const int m = m0 + wm * 32 + mi * 16 + h * 8 + r;
          out[(size_t)m * 1024 + o] = f2bf(acc[mi][j][r] + bo);
        }
      } else {
        // V transposed per head: Vt[(b*16+head)][hd][s], s contiguous
        const int srow = m0 + wm * 32 + mi * 16 + h * 8;   // s % 8 == 0
        const int bidx = srow >> 11;
        const int s    = srow & 2047;
        const int head = o >> 6, hd = o & 63;
        unsigned short t8[8];
#pragma unroll
        for (int r = 0; r < 8; ++r) t8[r] = f2bf(acc[mi][j][r] + bo);
        unsigned short* dst =
            Vt + (((size_t)(bidx * 16 + head) * 64 + hd) * 2048 + s);
        *(uint4*)dst = *(const uint4*)t8;
      }
    }
  }
}

// ---------------------------------------------------------------------------
// Fused attention, one wave per 16-query tile of one (b,head).
// 4*16*128 = 8192 waves = 1024 blocks x 8 waves.
__global__ void __launch_bounds__(256)
attn(const unsigned short* __restrict__ Qb,   // [8192][1024] bf16
     const unsigned short* __restrict__ Kb,   // [8192][1024] bf16
     const unsigned short* __restrict__ Vt,   // [64][64][2048] bf16
     float* __restrict__ out)                 // [4][2048][1024] f32
{
  __shared__ __align__(16) unsigned short plds[8][16 * 32];

  const int lane  = threadIdx.x & 31;
  const int wslot = threadIdx.x >> 5;
  const int wave  = blockIdx.x * 8 + wslot;
  if (wave >= 8192) return;
  const int qtile = wave & 127;
  const int head  = (wave >> 7) & 15;
  const int b     = wave >> 11;

  const int q0row = b * 2048 + qtile * 16;
  const int dcol  = head * 64;
  const int nlane = lane & 15, h = lane >> 4;

  const v16bf qa0 = load_frag_a(Qb, q0row, 1024, dcol,      lane);
  const v16bf qa1 = load_frag_a(Qb, q0row, 1024, dcol + 32, lane);

  v8f O[4] = {};
  float mrow[8], lrow[8];
#pragma unroll
  for (int r = 0; r < 8; ++r) { mrow[r] = -1e30f; lrow[r] = 0.0f; }

  const unsigned short* vbase = Vt + (size_t)(b * 16 + head) * 64 * 2048;
  unsigned short* myp = plds[wslot];
  const int krow0 = b * 2048;

  for (int t0 = 0; t0 < 2048; t0 += 32) {
    if (t0 + 32 < 2048) {   // warm L2->L0 for next key/value tiles
      __builtin_prefetch(Kb + (size_t)(krow0 + t0 + 32 + nlane) * 1024 + dcol, 0, 1);
      __builtin_prefetch(vbase + (size_t)(nlane) * 2048 + t0 + 32, 0, 1);
    }

    // ---- scores S[16q x 32t] = Q(16x64) . K^T : 4 WMMAs
    v8f s0 = {}, s1 = {};
    {
      v16bf kb00 = load_frag_b(Kb, krow0 + t0,      1024, dcol,      lane);
      v16bf kb01 = load_frag_b(Kb, krow0 + t0,      1024, dcol + 32, lane);
      s0 = __builtin_amdgcn_wmma_f32_16x16x32_bf16(false, qa0, false, kb00,
                                                   (short)0, s0, false, false);
      s0 = __builtin_amdgcn_wmma_f32_16x16x32_bf16(false, qa1, false, kb01,
                                                   (short)0, s0, false, false);
      v16bf kb10 = load_frag_b(Kb, krow0 + t0 + 16, 1024, dcol,      lane);
      v16bf kb11 = load_frag_b(Kb, krow0 + t0 + 16, 1024, dcol + 32, lane);
      s1 = __builtin_amdgcn_wmma_f32_16x16x32_bf16(false, qa0, false, kb10,
                                                   (short)0, s1, false, false);
      s1 = __builtin_amdgcn_wmma_f32_16x16x32_bf16(false, qa1, false, kb11,
                                                   (short)0, s1, false, false);
    }

    // ---- online softmax (rows span 16-lane halves; butterfly reduce)
    const float scale = 0.125f;               // 1/sqrt(64)
    float fac[8], p0v[8], p1v[8];
#pragma unroll
    for (int r = 0; r < 8; ++r) {
      const float a0 = s0[r] * scale, a1 = s1[r] * scale;
      float mx = fmaxf(a0, a1);
      mx = fmaxf(mx, __shfl_xor(mx, 1));
      mx = fmaxf(mx, __shfl_xor(mx, 2));
      mx = fmaxf(mx, __shfl_xor(mx, 4));
      mx = fmaxf(mx, __shfl_xor(mx, 8));
      const float mnew = fmaxf(mrow[r], mx);
      const float f    = __expf(mrow[r] - mnew);
      const float p0   = __expf(a0 - mnew);
      const float p1   = __expf(a1 - mnew);
      float rs = p0 + p1;
      rs += __shfl_xor(rs, 1);
      rs += __shfl_xor(rs, 2);
      rs += __shfl_xor(rs, 4);
      rs += __shfl_xor(rs, 8);
      lrow[r] = lrow[r] * f + rs;
      mrow[r] = mnew;
      fac[r] = f; p0v[r] = p0; p1v[r] = p1;
    }
#pragma unroll
    for (int j = 0; j < 4; ++j)
#pragma unroll
      for (int r = 0; r < 8; ++r) O[j][r] *= fac[r];

    // ---- re-layout P: C layout -> A layout via 1KB per-wave LDS tile
#pragma unroll
    for (int r = 0; r < 8; ++r) {
      const int m = r + 8 * h;
      myp[m * 32 + nlane]      = f2bf(p0v[r]);   // t = nlane
      myp[m * 32 + 16 + nlane] = f2bf(p1v[r]);   // t = 16 + nlane
    }
    asm volatile("s_wait_dscnt 0" ::: "memory"); // same-wave LDS RAW
    FragAB pa;
    {
      const unsigned short* rp = myp + nlane * 32 + h * 8;  // row m=nlane
      pa.q[0] = *(const uint4*)(rp);
      pa.q[1] = *(const uint4*)(rp + 16);
    }

    // ---- O[16q x 64d] += P(16x32) x V(32x64) : 4 WMMAs
#pragma unroll
    for (int j = 0; j < 4; ++j) {
      const unsigned short* vr =
          vbase + (size_t)(j * 16 + nlane) * 2048 + t0 + h * 8;
      FragAB vb_;
      vb_.q[0] = *(const uint4*)(vr);
      vb_.q[1] = *(const uint4*)(vr + 16);
      O[j] = __builtin_amdgcn_wmma_f32_16x16x32_bf16(
                 false, pa.v, false, vb_.v, (short)0, O[j], false, false);
    }
  }

  // ---- epilogue: divide by row sums, store f32 into [b,s,D]
#pragma unroll
  for (int j = 0; j < 4; ++j) {
#pragma unroll
    for (int r = 0; r < 8; ++r) {
      const int m = r + 8 * h;
      out[(size_t)(q0row + m) * 1024 + dcol + j * 16 + nlane] =
          O[j][r] / lrow[r];
    }
  }
}

// ---------------------------------------------------------------------------
extern "C" void kernel_launch(void* const* d_in, const int* in_sizes, int n_in,
                              void* d_out, int out_size, void* d_ws, size_t ws_size,
                              hipStream_t stream) {
  const float* x  = (const float*)d_in[0];
  const float* Wq = (const float*)d_in[1];
  const float* bq = (const float*)d_in[2];
  const float* Wk = (const float*)d_in[3];
  const float* bk = (const float*)d_in[4];
  const float* Wv = (const float*)d_in[5];
  const float* bv = (const float*)d_in[6];
  float* out = (float*)d_out;

  char* ws = (char*)d_ws;
  unsigned short* xb = (unsigned short*)(ws);                       // 16 MB
  unsigned short* wb = (unsigned short*)(ws + (16u << 20));         //  6 MB
  unsigned short* Qb = (unsigned short*)(ws + (22u << 20));         // 16 MB
  unsigned short* Kb = (unsigned short*)(ws + (38u << 20));         // 16 MB
  unsigned short* Vt = (unsigned short*)(ws + (54u << 20));         // 16 MB

  cvt_f32_bf16<<<4096, 256, 0, stream>>>(x,  xb,              8388608);
  cvt_f32_bf16<<<1024, 256, 0, stream>>>(Wq, wb,              1048576);
  cvt_f32_bf16<<<1024, 256, 0, stream>>>(Wk, wb + (1u << 20), 1048576);
  cvt_f32_bf16<<<1024, 256, 0, stream>>>(Wv, wb + (2u << 20), 1048576);

  qkv_proj<<<1536, 256, 0, stream>>>(xb, wb, bq, bk, bv, Qb, Kb, Vt);
  attn<<<1024, 256, 0, stream>>>(Qb, Kb, Vt, out);
}
